// Attention_11991548690893
// MI455X (gfx1250) — matrix operand order repaired and checked by reading, not compiled, not run on hardware
//
#include <hip/hip_runtime.h>
#include <cstdint>
#include <cstddef>

// ---------------------------------------------------------------------------
// CDNA5 (gfx1250) WMMA types
// ---------------------------------------------------------------------------
typedef __attribute__((ext_vector_type(16))) __bf16 v16bf;
typedef __attribute__((ext_vector_type(8)))  float  v8f;
typedef __attribute__((ext_vector_type(4)))  int    v4i;

union Frag16 {
    v16bf v;
    unsigned int u[8];
    unsigned short s[16];
};

__device__ __forceinline__ unsigned short f32_to_bf16(float f) {
    unsigned int x = __float_as_uint(f);
    unsigned int r = (x + 0x7FFFu + ((x >> 16) & 1u)) >> 16;   // RNE
    return (unsigned short)r;
}

// ---------------------------------------------------------------------------
// Async global -> LDS staging (CDNA5 ASYNCcnt path), with safe fallback
// ---------------------------------------------------------------------------
#define GAS __attribute__((address_space(1)))
#define LAS __attribute__((address_space(3)))

#if __has_builtin(__builtin_amdgcn_global_load_async_to_lds_b128)
#define HAS_ASYNC_LDS 1
#else
#define HAS_ASYNC_LDS 0
#endif

__device__ __forceinline__ void stage16(void* lds, const void* g) {
#if HAS_ASYNC_LDS
    __builtin_amdgcn_global_load_async_to_lds_b128(
        (GAS v4i*)g, (LAS v4i*)lds, 0, 0);
#else
    *(uint4*)lds = *(const uint4*)g;
#endif
}

// Wait until at most N async-LDS loads remain outstanding (loads complete in
// order, so N == loads-in-flight-for-next-tile keeps the pipeline running).
template <int N>
__device__ __forceinline__ void stage_wait_n() {
#if HAS_ASYNC_LDS
#if __has_builtin(__builtin_amdgcn_s_wait_asynccnt)
    __builtin_amdgcn_s_wait_asynccnt(N);
#else
    asm volatile("s_wait_asynccnt %0" :: "i"(N) : "memory");
#endif
#endif
}

// ---------------------------------------------------------------------------
// Problem constants
// ---------------------------------------------------------------------------
static constexpr int D      = 1024;  // d_model
static constexpr int NH     = 16;    // heads
static constexpr int DH     = 64;    // head dim
static constexpr int LSEQ   = 2048;  // sequence length
static constexpr int NB     = 2;     // batch
static constexpr int MROWS  = NB * LSEQ;  // 4096

// GEMM tiling
static constexpr int BM = 128, BN = 64, BK = 32;

// ---------------------------------------------------------------------------
// f32 -> bf16 conversion kernels
// ---------------------------------------------------------------------------
__global__ void convert_f32_bf16(const float* __restrict__ src,
                                 unsigned short* __restrict__ dst, int n) {
    int i = blockIdx.x * blockDim.x + threadIdx.x;
    if (i < n) dst[i] = f32_to_bf16(src[i]);
}

// K-slice of W_attn (cols D..2D), TRANSPOSED: dst[n][k] = W_attn[k][D+n]
__global__ void convert_wkT(const float* __restrict__ W_attn,
                            unsigned short* __restrict__ dst) {
    int i = blockIdx.x * blockDim.x + threadIdx.x;  // over 1024*1024
    int n = i >> 10, k = i & 1023;
    dst[i] = f32_to_bf16(W_attn[k * (3 * D) + D + n]);
}

// W_proj TRANSPOSED: dst[n][k] = W_proj[k][n]
__global__ void convert_wpT(const float* __restrict__ W_proj,
                            unsigned short* __restrict__ dst) {
    int i = blockIdx.x * blockDim.x + threadIdx.x;  // over 1024*1024
    int n = i >> 10, k = i & 1023;
    dst[i] = f32_to_bf16(W_proj[k * D + n]);
}

// ---------------------------------------------------------------------------
// Tiled bf16 WMMA GEMM with double-buffered async LDS staging:
//   C(4096x1024) = A(4096x1024) * BT^T + bias
//   A  : row-major [m][k]
//   BT : TRANSPOSED weights [n][k]  -> B fragments are contiguous-K reads
// mode 0: f32 output, row-major
// mode 1: bf16 output to kh layout [b][h][l][dh] AND khT layout [b][h][dh][l]
// ---------------------------------------------------------------------------
__global__ __launch_bounds__(256) void gemm_bf16(
    const unsigned short* __restrict__ A, const unsigned short* __restrict__ BT,
    const float* __restrict__ bias, float* __restrict__ outF,
    unsigned short* __restrict__ outH, unsigned short* __restrict__ outHT,
    int mode)
{
    __shared__ __align__(16) unsigned short As [2][BM * BK];  // 2 x 8 KB
    __shared__ __align__(16) unsigned short BsT[2][BN * BK];  // 2 x 4 KB

    const int tid  = threadIdx.x;
    const int lane = tid & 31;
    const int wave = tid >> 5;
    const int hl   = lane >> 4;      // half-wave (0/1)
    const int nl   = lane & 15;
    const int m0   = blockIdx.y * BM;
    const int n0   = blockIdx.x * BN;
    const int wm0  = (wave & 3) * 32;   // 4 waves along M
    const int wn0  = (wave >> 2) * 32;  // 2 waves along N

    // 3 async b128 per thread per tile
    auto stage_tile = [&](int k0, int p) {
        #pragma unroll
        for (int i = 0; i < 2; ++i) {
            int u = tid + i * 256;
            int row = u >> 2, off = (u & 3) * 8;
            stage16(&As[p][row * BK + off], &A[(size_t)(m0 + row) * D + k0 + off]);
        }
        int row = tid >> 2, off = (tid & 3) * 8;
        stage16(&BsT[p][row * BK + off], &BT[(size_t)(n0 + row) * D + k0 + off]);
    };

    v8f acc[2][2] = {};

    stage_tile(0, 0);   // prologue: tile 0 -> buffer 0

    for (int it = 0; it < D / BK; ++it) {
        const int k0 = it * BK;
        const int p  = it & 1;
        if (k0 + BK < D) {
            stage_tile(k0 + BK, p ^ 1);   // next tile in flight under compute
            stage_wait_n<3>();            // current tile resident; next stays async
            if (k0 + 2 * BK < D) {        // warm L2 for tile i+2
                __builtin_prefetch(&A [(size_t)(m0 + (tid & 127)) * D + k0 + 2 * BK], 0, 0);
                __builtin_prefetch(&BT[(size_t)(n0 + (tid &  63)) * D + k0 + 2 * BK], 0, 0);
            }
        } else {
            stage_wait_n<0>();
        }
        __syncthreads();

        // A fragments: lane M=nl; VGPR j -> K = (j>=4?16:0) + 8*hl + 2*(j&3)
        //  -> two contiguous 16B runs per lane (ds_load_b128 x2)
        Frag16 af[2];
        #pragma unroll
        for (int sm = 0; sm < 2; ++sm) {
            int ml = wm0 + sm * 16 + nl;
            #pragma unroll
            for (int j = 0; j < 8; ++j) {
                int kk = ((j & 4) ? 16 : 0) + hl * 8 + 2 * (j & 3);
                af[sm].u[j] = *(const unsigned int*)&As[p][ml * BK + kk];
            }
        }
        // B fragments: lane N=nl; VGPR j -> K = (j>=4?16:0) + 8*hl + 2*(j&3)
        //  -> two contiguous 16B runs per lane (ds_load_b128 x2)
        Frag16 bfr[2];
        #pragma unroll
        for (int sn = 0; sn < 2; ++sn) {
            int nn = wn0 + sn * 16 + nl;
            #pragma unroll
            for (int j = 0; j < 8; ++j) {
                int kk = ((j & 4) ? 16 : 0) + hl * 8 + 2 * (j & 3);
                bfr[sn].u[j] = *(const unsigned int*)&BsT[p][nn * BK + kk];
            }
        }
        #pragma unroll
        for (int sm = 0; sm < 2; ++sm)
            #pragma unroll
            for (int sn = 0; sn < 2; ++sn)
                acc[sm][sn] = __builtin_amdgcn_wmma_f32_16x16x32_bf16(
                    false, af[sm].v, false, bfr[sn].v,
                    (short)0, acc[sm][sn], false, false);

        __syncthreads();   // all waves done reading buf[p] before it is restaged
    }

    // Epilogue: C layout — VGPR r -> M = r + 8*hl, lane -> N = nl
    #pragma unroll
    for (int sm = 0; sm < 2; ++sm)
        #pragma unroll
        for (int sn = 0; sn < 2; ++sn)
            #pragma unroll
            for (int r = 0; r < 8; ++r) {
                int gm = m0 + wm0 + sm * 16 + r + 8 * hl;
                int gn = n0 + wn0 + sn * 16 + nl;
                float val = acc[sm][sn][r] + bias[gn];
                if (mode == 0) {
                    outF[(size_t)gm * D + gn] = val;
                } else {
                    int b = gm >> 11, ll = gm & (LSEQ - 1);
                    int h = gn >> 6,  dh = gn & (DH - 1);
                    unsigned short hv = f32_to_bf16(val);
                    outH [(((size_t)b * NH + h) * LSEQ + ll) * DH + dh] = hv;
                    outHT[(((size_t)b * NH + h) * DH + dh) * LSEQ + ll] = hv;
                }
            }
}

// ---------------------------------------------------------------------------
// Flash-style self-attention (Q = K = kh), one wave per 16 query rows,
// double-buffered async key-block staging.
//   kh : bf16 [b][h][l][dh]   (S-step operand, contiguous dh)
//   khT: bf16 [b][h][dh][l]   (O-step operand, contiguous l)
//   out: bf16 [b][l][d]       (d = h*64+dh)
// ---------------------------------------------------------------------------
__global__ __launch_bounds__(32) void attn_kernel(
    const unsigned short* __restrict__ kh, const unsigned short* __restrict__ khT,
    unsigned short* __restrict__ attnout)
{
    __shared__ __align__(16) unsigned short Ks [2][32 * DH];  // [key][dh], 2 x 4 KB
    __shared__ __align__(16) unsigned short KsT[2][DH * 32];  // [dh][key], 2 x 4 KB
    __shared__ __align__(16) unsigned short Ps [16 * 32];     // P tile,    1 KB

    const int lane = threadIdx.x;
    const int hl   = lane >> 4;
    const int nl   = lane & 15;
    const int blk  = blockIdx.x;       // 0..4095
    const int qb   = blk & 127;        // 128 q-blocks of 16 rows
    const int bh   = blk >> 7;         // b*NH + h
    const int q0   = qb * 16;
    const unsigned short* kp  = kh  + (size_t)bh * LSEQ * DH;
    const unsigned short* kpT = khT + (size_t)bh * DH * LSEQ;

    // 16 async b128 per lane per key block
    auto stage_kv = [&](int k0, int p) {
        #pragma unroll
        for (int i = 0; i < 8; ++i) {
            int u = lane + i * 32;
            int row = u >> 3, off = (u & 7) * 8;   // Ks: 8 x16B per row
            stage16(&Ks[p][row * DH + off], &kp[(size_t)(k0 + row) * DH + off]);
            int rowT = u >> 2, offT = (u & 3) * 8; // KsT: 4 x16B per row
            stage16(&KsT[p][rowT * 32 + offT], &kpT[(size_t)rowT * LSEQ + k0 + offT]);
        }
    };

    // Q fragments (16x64 -> two 16x32 A fragments), kept in registers
    Frag16 qa, qc;
    #pragma unroll
    for (int j = 0; j < 8; ++j) {
        int kk = ((j & 4) ? 16 : 0) + hl * 8 + 2 * (j & 3);
        qa.u[j] = *(const unsigned int*)&kp[(size_t)(q0 + nl) * DH + kk];
        qc.u[j] = *(const unsigned int*)&kp[(size_t)(q0 + nl) * DH + 32 + kk];
    }

    float mi[8], li[8];
    #pragma unroll
    for (int r = 0; r < 8; ++r) { mi[r] = -3.0e38f; li[r] = 0.0f; }
    v8f O[4] = {};

    stage_kv(0, 0);   // prologue

    for (int it = 0; it < LSEQ / 32; ++it) {
        const int k0 = it * 32;
        const int p  = it & 1;
        if (k0 + 32 < LSEQ) {
            stage_kv(k0 + 32, p ^ 1);   // next key block in flight under compute
            stage_wait_n<16>();
            if (k0 + 64 < LSEQ)
                __builtin_prefetch(&kp[(size_t)(k0 + 64 + (lane & 31)) * DH], 0, 0);
        } else {
            stage_wait_n<0>();
        }
        __syncthreads();

        // S = Q * K^T : B[k=dh][n=key] = Ks[n][k]  (contiguous dh -> b128)
        v8f S[2];
        #pragma unroll
        for (int nt = 0; nt < 2; ++nt) {
            Frag16 b0, b1;
            int n = nt * 16 + nl;
            #pragma unroll
            for (int j = 0; j < 8; ++j) {
                int kk = ((j & 4) ? 16 : 0) + hl * 8 + 2 * (j & 3);
                b0.u[j] = *(const unsigned int*)&Ks[p][n * DH + kk];
                b1.u[j] = *(const unsigned int*)&Ks[p][n * DH + 32 + kk];
            }
            v8f s = {};
            s = __builtin_amdgcn_wmma_f32_16x16x32_bf16(false, qa.v, false, b0.v,
                                                        (short)0, s, false, false);
            s = __builtin_amdgcn_wmma_f32_16x16x32_bf16(false, qc.v, false, b1.v,
                                                        (short)0, s, false, false);
            S[nt] = s;
        }
        #pragma unroll
        for (int nt = 0; nt < 2; ++nt)
            #pragma unroll
            for (int r = 0; r < 8; ++r)
                S[nt][r] *= 0.125f;  // 1/sqrt(64)

        // Online softmax: row reductions across the 16-lane half groups
        float rmax[8], rsum[8], alpha[8];
        #pragma unroll
        for (int r = 0; r < 8; ++r) rmax[r] = fmaxf(S[0][r], S[1][r]);
        #pragma unroll
        for (int mk = 1; mk <= 8; mk <<= 1)
            #pragma unroll
            for (int r = 0; r < 8; ++r)
                rmax[r] = fmaxf(rmax[r], __shfl_xor(rmax[r], mk, 32));
        #pragma unroll
        for (int r = 0; r < 8; ++r) {
            float mn = fmaxf(mi[r], rmax[r]);
            alpha[r] = __expf(mi[r] - mn);
            mi[r] = mn;
        }
        #pragma unroll
        for (int nt = 0; nt < 2; ++nt)
            #pragma unroll
            for (int r = 0; r < 8; ++r)
                S[nt][r] = __expf(S[nt][r] - mi[r]);
        #pragma unroll
        for (int r = 0; r < 8; ++r) rsum[r] = S[0][r] + S[1][r];
        #pragma unroll
        for (int mk = 1; mk <= 8; mk <<= 1)
            #pragma unroll
            for (int r = 0; r < 8; ++r)
                rsum[r] += __shfl_xor(rsum[r], mk, 32);
        #pragma unroll
        for (int r = 0; r < 8; ++r) li[r] = li[r] * alpha[r] + rsum[r];
        #pragma unroll
        for (int t = 0; t < 4; ++t)
            #pragma unroll
            for (int r = 0; r < 8; ++r)
                O[t][r] *= alpha[r];

        // C-layout P -> LDS (bf16) -> A-layout reload
        #pragma unroll
        for (int r = 0; r < 8; ++r) {
            Ps[(r + 8 * hl) * 32 + nl]      = f32_to_bf16(S[0][r]);
            Ps[(r + 8 * hl) * 32 + 16 + nl] = f32_to_bf16(S[1][r]);
        }
        __syncthreads();

        Frag16 pa;
        #pragma unroll
        for (int j = 0; j < 8; ++j) {
            int kk = ((j & 4) ? 16 : 0) + hl * 8 + 2 * (j & 3);
            pa.u[j] = *(const unsigned int*)&Ps[nl * 32 + kk];
        }
        // O += P * Kblock : B[k=key][n=dh] = KsT[n][k] (contiguous key -> b128)
        #pragma unroll
        for (int t = 0; t < 4; ++t) {
            Frag16 bb;
            #pragma unroll
            for (int j = 0; j < 8; ++j) {
                int kk = ((j & 4) ? 16 : 0) + hl * 8 + 2 * (j & 3);
                bb.u[j] = *(const unsigned int*)&KsT[p][(t * 16 + nl) * 32 + kk];
            }
            O[t] = __builtin_amdgcn_wmma_f32_16x16x32_bf16(false, pa.v, false, bb.v,
                                                           (short)0, O[t], false, false);
        }

        __syncthreads();   // done reading Ks/KsT[p] and Ps before restaging
    }

    // Finalize: divide by row sums, write bf16 [b][l][d]
    const int b = bh >> 4, h = bh & 15;
    #pragma unroll
    for (int r = 0; r < 8; ++r) {
        float inv = 1.0f / li[r];
        int lq = q0 + r + 8 * hl;
        #pragma unroll
        for (int t = 0; t < 4; ++t) {
            int gd = h * DH + t * 16 + nl;
            attnout[((size_t)b * LSEQ + lq) * D + gd] = f32_to_bf16(O[t][r] * inv);
        }
    }
}

// ---------------------------------------------------------------------------
// Launch
// ---------------------------------------------------------------------------
extern "C" void kernel_launch(void* const* d_in, const int* in_sizes, int n_in,
                              void* d_out, int out_size, void* d_ws, size_t ws_size,
                              hipStream_t stream)
{
    (void)in_sizes; (void)n_in; (void)out_size; (void)ws_size;

    const float* x      = (const float*)d_in[0];
    const float* W_attn = (const float*)d_in[1];
    const float* b_attn = (const float*)d_in[2];
    const float* W_proj = (const float*)d_in[3];
    const float* b_proj = (const float*)d_in[4];
    float* out = (float*)d_out;

    // Workspace layout (bytes)
    char* ws = (char*)d_ws;
    unsigned short* xb  = (unsigned short*)(ws);             // 4096x1024 bf16   ( 8 MB)
    unsigned short* wkT = (unsigned short*)(ws +  8388608);  // 1024x1024 bf16 T ( 2 MB)
    unsigned short* wpT = (unsigned short*)(ws + 10485760);  // 1024x1024 bf16 T ( 2 MB)
    unsigned short* khb = (unsigned short*)(ws + 12582912);  // [2][16][2048][64] ( 8 MB)
    unsigned short* khT = (unsigned short*)(ws + 20971520);  // [2][16][64][2048] ( 8 MB)
    unsigned short* ao  = (unsigned short*)(ws + 29360128);  // 4096x1024 bf16   ( 8 MB)

    convert_f32_bf16<<<(MROWS * D) / 256, 256, 0, stream>>>(x, xb, MROWS * D);
    convert_wkT<<<(D * D) / 256, 256, 0, stream>>>(W_attn, wkT);
    convert_wpT<<<(D * D) / 256, 256, 0, stream>>>(W_proj, wpT);

    // kh = x @ W_k + b_k, stored as [b][h][l][dh] and [b][h][dh][l] bf16
    gemm_bf16<<<dim3(D / BN, MROWS / BM), 256, 0, stream>>>(
        xb, wkT, b_attn + D, nullptr, khb, khT, 1);

    // self-attention
    attn_kernel<<<NB * NH * (LSEQ / 16), 32, 0, stream>>>(khb, khT, ao);

    // out = attnout @ W_proj + b_proj  (f32)
    gemm_bf16<<<dim3(D / BN, MROWS / BM), 256, 0, stream>>>(
        ao, wpT, b_proj, out, nullptr, nullptr, 0);
}
